// FreeItemKNN_46179488367358
// MI455X (gfx1250) — matrix operand order repaired and checked by reading, not compiled, not run on hardware
//
#include <hip/hip_runtime.h>
#include <hip/hip_bf16.h>

// FreeItemKNN reformulated as   R = W @ M,  logits[b,c] = R[item[b,c], b]
//   W : 10000 x 10000 fp32, streamed from HBM exactly once (~400MB -> ~17us @ 23.3TB/s)
//   M : per-batch masked seen-item counts, stored transposed bf16 (L2 resident)
//   R : 10000 x 128 fp32 (L2 resident)
// GEMM: v_wmma_f32_16x16x32_bf16 with a hi/lo bf16 split of W (~2^-18 rel error).
// The split is done ONCE per 16-row tile by the whole workgroup into LDS
// (double-buffered); A fragments are additionally double-buffered in registers
// so ds_load latency pipelines across K-steps instead of s_wait_dscnt 0 stalls.

#define NUM_ITEMS 10000
#define BATCH     128
#define N_CAND    10000
#define N_SEEN    100
#define KDIM      10000
#define CHUNK     256
#define NCHUNK    40
#define KPAD      (CHUNK * NCHUNK)   // 10240, Mt rows zero-padded to this
#define AROWU4    33                 // uint4 per LDS row: 32 data + 1 pad (bank spread)

typedef __bf16 bf16;
typedef __attribute__((ext_vector_type(16))) __bf16 v16bf;
typedef __attribute__((ext_vector_type(8)))  float  v8f;

// ---------------------------------------------------------------------------
// Kernel 1: Mt[b][j] = bf16(# unmasked occurrences of j in seen[b,:]);
// "masked" = j appears anywhere in item[b,:]. One block per batch.
// ---------------------------------------------------------------------------
__global__ __launch_bounds__(256) void build_mt(const int* __restrict__ item,
                                                const int* __restrict__ seen,
                                                bf16* __restrict__ Mt) {
  __shared__ unsigned int bitmap[(NUM_ITEMS + 31) / 32];  // 313 dwords
  __shared__ unsigned int cnt[NUM_ITEMS];                 // 40KB
  const int b = blockIdx.x;
  const int t = threadIdx.x;

  for (int i = t; i < (NUM_ITEMS + 31) / 32; i += 256) bitmap[i] = 0u;
  for (int i = t; i < NUM_ITEMS; i += 256) cnt[i] = 0u;
  __syncthreads();

  const int* itemb = item + (size_t)b * N_CAND;
  for (int i = t; i < N_CAND; i += 256) {
    int j = itemb[i];
    atomicOr(&bitmap[j >> 5], 1u << (j & 31));
  }
  __syncthreads();

  if (t < N_SEEN) {
    int j = seen[b * N_SEEN + t];
    if (!((bitmap[j >> 5] >> (j & 31)) & 1u)) atomicAdd(&cnt[j], 1u);
  }
  __syncthreads();

  bf16* row = Mt + (size_t)b * KPAD;
  for (int i = t; i < KPAD; i += 256) {
    row[i] = (i < NUM_ITEMS) ? (bf16)(float)cnt[i] : (bf16)0.0f;
  }
}

// ---------------------------------------------------------------------------
// Kernel 2: R[10000 x 128] = W @ M via WMMA bf16, LDS-staged + double-buffered.
// 625 blocks (16-row tiles), 256 threads = 8 wave32s; wave w owns batch cols
// [16w, 16w+16). Per K-chunk of 256: cooperative load/split/pack of the W tile
// into LDS once, then each wave runs 8 K-steps of 2 WMMAs from ds_load_b128
// fragments. W bounds (K tail 10000..10239) handled in staging only.
// ---------------------------------------------------------------------------
__global__ __launch_bounds__(256) void gemm_wmma(const float* __restrict__ W,
                                                 const bf16* __restrict__ Mt,
                                                 float* __restrict__ R) {
  __shared__ uint4 Ahi[2][16 * AROWU4];   // 16 rows x 256 bf16, padded rows
  __shared__ uint4 Alo[2][16 * AROWU4];

  const int t       = threadIdx.x;
  const int lane    = t & 31;
  const int wid     = t >> 5;             // 0..7 -> batch-column tile
  const int rowbase = blockIdx.x * 16;    // 625 * 16 = 10000 rows exactly

  // Staging role: thread t -> W row (t>>4), 16 consecutive k at ((t&15)*16).
  const int sr = t >> 4;
  const int sk = (t & 15) * 16;
  const float* Wrow = W + (size_t)(rowbase + sr) * KDIM + sk;   // 16B aligned
  const int sidx = sr * AROWU4 + (sk >> 3);                     // 2 uint4 slots

  // Compute role (A frag): lanes 0-15 row=lane, K {0..15};
  // lanes 16-31 same rows, K {16..31}. B: lane holds col lane&15,
  // K = (lane>>4)*16 + {0..15}.
  const int half = lane >> 4;
  const int ar   = lane & 15;
  const int bcol = wid * 16 + ar;
  const bf16* Mp = Mt + (size_t)bcol * KPAD + half * 16;        // 32B aligned
  const int  fb  = ar * AROWU4 + 2 * half;                      // frag uint4 base

  float f[16];
  v8f acc_hi = {}, acc_lo = {};

  auto load_regs = [&](int c) {
    const int kbase = c * CHUNK + sk;
    #pragma unroll
    for (int i = 0; i < 16; i += 4) {
      if (kbase + i < KDIM) {                  // whole float4 valid or whole OOB
        const float4 v = *(const float4*)(Wrow + c * CHUNK + i);
        f[i] = v.x; f[i + 1] = v.y; f[i + 2] = v.z; f[i + 3] = v.w;
      } else {
        f[i] = 0.f; f[i + 1] = 0.f; f[i + 2] = 0.f; f[i + 3] = 0.f;
      }
    }
  };

  auto cvt_store = [&](int buf) {
    #pragma unroll
    for (int g = 0; g < 2; ++g) {
      union { __bf16 h[8]; uint4 u; } ph, pl;
      #pragma unroll
      for (int i = 0; i < 8; ++i) {
        float x = f[g * 8 + i];
        __bf16 h = (__bf16)x;                  // RNE
        ph.h[i] = h;
        pl.h[i] = (__bf16)(x - (float)h);      // exact residual, then RNE
      }
      Ahi[buf][sidx + g] = ph.u;               // ds_store_b128
      Alo[buf][sidx + g] = pl.u;
    }
  };

  auto compute = [&](int c, int buf) {
    const bf16* mp = Mp + c * CHUNK;
    union F { uint4 u[2]; v16bf v; };
    F ah[2], al[2];
    // Preload step-0 fragments; thereafter prefetch s+1 into the other
    // register set before issuing step s's WMMAs (pipelines ds latency).
    ah[0].u[0] = Ahi[buf][fb + 0];
    ah[0].u[1] = Ahi[buf][fb + 1];
    al[0].u[0] = Alo[buf][fb + 0];
    al[0].u[1] = Alo[buf][fb + 1];
    #pragma unroll
    for (int s = 0; s < CHUNK / 32; ++s) {     // 8 K-steps of 32
      const int cur = s & 1;
      const int nxt = cur ^ 1;
      if (s + 1 < CHUNK / 32) {
        ah[nxt].u[0] = Ahi[buf][fb + 4 * (s + 1)];
        ah[nxt].u[1] = Ahi[buf][fb + 4 * (s + 1) + 1];
        al[nxt].u[0] = Alo[buf][fb + 4 * (s + 1)];
        al[nxt].u[1] = Alo[buf][fb + 4 * (s + 1) + 1];
      }
      v16bf bb = *(const v16bf*)(mp + s * 32); // 2x global_load_b128 (L2)
      acc_hi = __builtin_amdgcn_wmma_f32_16x16x32_bf16(false, ah[cur].v, false, bb,
                                                       (short)0, acc_hi, false, false);
      acc_lo = __builtin_amdgcn_wmma_f32_16x16x32_bf16(false, al[cur].v, false, bb,
                                                       (short)0, acc_lo, false, false);
    }
  };

  // Software pipeline: global loads for c+1 issue before compute(c);
  // split+LDS stores after; one barrier per chunk; double-buffered LDS.
  load_regs(0);
  cvt_store(0);
  __syncthreads();
  #pragma unroll 1
  for (int c = 0; c < NCHUNK; ++c) {
    const int buf = c & 1;
    if (c + 1 < NCHUNK) load_regs(c + 1);
    compute(c, buf);
    if (c + 1 < NCHUNK) cvt_store(buf ^ 1);
    __syncthreads();
  }

  // D layout: VGPR v, lane L -> row = rowbase + v + 8*(L>=16), col = bcol
  const int drow0 = rowbase + 8 * half;
  #pragma unroll
  for (int v = 0; v < 8; ++v) {
    R[(size_t)(drow0 + v) * BATCH + bcol] = acc_hi[v] + acc_lo[v];
  }
}

// ---------------------------------------------------------------------------
// Kernel 3: out[b,c] = R[item[b,c], b]   (R is 5MB -> pure L2 gather)
// ---------------------------------------------------------------------------
__global__ __launch_bounds__(256) void gather_out(const int* __restrict__ item,
                                                  const float* __restrict__ R,
                                                  float* __restrict__ out) {
  int idx = blockIdx.x * 256 + threadIdx.x;
  if (idx >= BATCH * N_CAND) return;
  int b = idx / N_CAND;
  int j = item[idx];
  out[idx] = R[(size_t)j * BATCH + b];
}

extern "C" void kernel_launch(void* const* d_in, const int* in_sizes, int n_in,
                              void* d_out, int out_size, void* d_ws, size_t ws_size,
                              hipStream_t stream) {
  (void)in_sizes; (void)n_in; (void)out_size; (void)ws_size;
  // setup_inputs order: user (unused), item, seen_items, weights
  const int*   item = (const int*)d_in[1];
  const int*   seen = (const int*)d_in[2];
  const float* W    = (const float*)d_in[3];
  float*       out  = (float*)d_out;

  bf16*  Mt = (bf16*)d_ws;                                                   // 2,621,440 B
  float* R  = (float*)((char*)d_ws + (size_t)BATCH * KPAD * sizeof(bf16));   // +5,120,000 B

  build_mt  <<<BATCH, 256, 0, stream>>>(item, seen, Mt);
  gemm_wmma <<<NUM_ITEMS / 16, 256, 0, stream>>>(W, Mt, R);
  gather_out<<<(BATCH * N_CAND + 255) / 256, 256, 0, stream>>>(item, R, out);
}
